// MultiModalTransformerForClassification_22969485099552
// MI455X (gfx1250) — matrix operand order repaired and checked by reading, not compiled, not run on hardware
//
#include <hip/hip_runtime.h>
#include <hip/hip_bf16.h>

typedef _Float16 v16h __attribute__((ext_vector_type(16)));
typedef float    v8f  __attribute__((ext_vector_type(8)));
typedef _Float16 h4   __attribute__((ext_vector_type(4)));
typedef _Float16 h8   __attribute__((ext_vector_type(8)));
typedef float    f4   __attribute__((ext_vector_type(4)));

// ---------------------------------------------------------------------------
// Generic batched + headed strided GEMM using CDNA5 WMMA f32_16x16x32_f16.
// C[b,h] = act( alpha * A[b,h] @ B[b,h](^T) + bias )
// Block: 128 threads = 4 waves; tile 64(M) x 64(N), K-step 64 (2 WMMA slices).
// LDS layouts are fragment-friendly: As[m][k], BsT[n][k] -> both A and B
// fragments are 2x contiguous 16B reads (ds_load_b128), conflict-free
// (row stride 144B = 36 dwords; 36*l mod 64 distinct for l in [0,16)).
// ---------------------------------------------------------------------------
__global__ __launch_bounds__(128)
void gemm_wmma_kernel(const float* __restrict__ A, const float* __restrict__ Bm,
                      const float* __restrict__ bias, float* __restrict__ C,
                      int M, int N, int K, int nheads,
                      long long sAb, long long sAh, int lda,
                      long long sBb, long long sBh, int ldb, int transB,
                      long long sCb, long long sCh, int ldc,
                      float alpha, int act)
{
    __shared__ _Float16 As [64][72];  // As[m][k],  64 x 64 (+8 pad)
    __shared__ _Float16 BsT[64][72];  // BsT[n][k], 64 x 64 (+8 pad)

    const int z  = blockIdx.z;
    const int bb = z / nheads;
    const int hh = z - bb * nheads;
    const float* Ap = A  + (long long)bb * sAb + (long long)hh * sAh;
    const float* Bp = Bm + (long long)bb * sBb + (long long)hh * sBh;
    float*       Cp = C  + (long long)bb * sCb + (long long)hh * sCh;

    const int mBase = blockIdx.y * 64;
    const int nBase = blockIdx.x * 64;
    const int tid   = threadIdx.x;
    const int wave  = tid >> 5;
    const int lane  = tid & 31;
    const int mW    = (wave >> 1) * 32;
    const int nW    = (wave & 1) * 32;
    const int l16   = lane & 15;
    const int aKoff = (lane >> 4) * 8;    // A-frag K sub-offset per half-wave
    const int bKoff = (lane >> 4) * 8;    // B-frag K sub-offset per half-wave

    const bool fullM = (mBase + 64 <= M);
    const bool fullN = (nBase + 64 <= N);
    const bool ldaV  = ((lda & 3) == 0);
    const bool ldbV  = ((ldb & 3) == 0);

    v8f acc[2][2];
#pragma unroll
    for (int i = 0; i < 2; ++i)
#pragma unroll
        for (int j = 0; j < 2; ++j)
#pragma unroll
            for (int r = 0; r < 8; ++r) acc[i][j][r] = 0.0f;

    for (int k0 = 0; k0 < K; k0 += 64) {
        const bool fullK = (k0 + 64 <= K);

        // ---------------- A tile: As[m][k], 64 x 64 ----------------
        if (fullM && fullK && ldaV) {
            for (int i = tid; i < 64 * 16; i += 128) {     // 8 f4 loads / thread
                int r = i >> 4, c4 = (i & 15) << 2;
                f4 v = *(const f4*)(Ap + (long long)(mBase + r) * lda + k0 + c4);
                h4 h;
                h[0] = (_Float16)v[0]; h[1] = (_Float16)v[1];
                h[2] = (_Float16)v[2]; h[3] = (_Float16)v[3];
                *(h4*)&As[r][c4] = h;
            }
        } else {
            // branchless: clamped address, unconditional load, select-zero
            for (int i = tid; i < 64 * 64; i += 128) {
                int r = i >> 6, c = i & 63;
                int gm = mBase + r, gk = k0 + c;
                int cm = (gm < M) ? gm : (M - 1);
                int ck = (gk < K) ? gk : (K - 1);
                float v = Ap[(long long)cm * lda + ck];
                As[r][c] = (_Float16)((gm < M && gk < K) ? v : 0.0f);
            }
        }

        // ---------------- B tile: BsT[n][k], 64 x 64 ----------------
        if (transB) {
            if (fullN && fullK && ldbV) {
                // global contiguous along k, LDS contiguous along k: both vector
                for (int i = tid; i < 64 * 16; i += 128) {
                    int n = i >> 4, k4 = (i & 15) << 2;
                    f4 v = *(const f4*)(Bp + (long long)(nBase + n) * ldb + k0 + k4);
                    h4 h;
                    h[0] = (_Float16)v[0]; h[1] = (_Float16)v[1];
                    h[2] = (_Float16)v[2]; h[3] = (_Float16)v[3];
                    *(h4*)&BsT[n][k4] = h;
                }
            } else {
                for (int i = tid; i < 64 * 64; i += 128) {
                    int kk = i >> 6, n = i & 63;
                    int gk = k0 + kk, gn = nBase + n;
                    int cn = (gn < N) ? gn : (N - 1);
                    int ck = (gk < K) ? gk : (K - 1);
                    float v = Bp[(long long)cn * ldb + ck];
                    BsT[n][kk] = (_Float16)((gk < K && gn < N) ? v : 0.0f);
                }
            }
        } else {
            if (fullN && fullK && ldbV) {
                // register 4x4 transpose: 4 f4 loads (k-major) -> 4 h4 stores (n-major)
                for (int i = tid; i < 256; i += 128) {     // 2 blocks / thread
                    int bn = (i & 15) << 2;                // n block
                    int bk = (i >> 4) << 2;                // k block
                    const float* src = Bp + (long long)(k0 + bk) * ldb + nBase + bn;
                    f4 r0 = *(const f4*)(src);
                    f4 r1 = *(const f4*)(src + ldb);
                    f4 r2 = *(const f4*)(src + 2 * (long long)ldb);
                    f4 r3 = *(const f4*)(src + 3 * (long long)ldb);
#pragma unroll
                    for (int c = 0; c < 4; ++c) {
                        h4 h;
                        h[0] = (_Float16)r0[c]; h[1] = (_Float16)r1[c];
                        h[2] = (_Float16)r2[c]; h[3] = (_Float16)r3[c];
                        *(h4*)&BsT[bn + c][bk] = h;
                    }
                }
            } else {
                for (int i = tid; i < 64 * 64; i += 128) {
                    int kk = i >> 6, n = i & 63;
                    int gk = k0 + kk, gn = nBase + n;
                    int cn = (gn < N) ? gn : (N - 1);
                    int ck = (gk < K) ? gk : (K - 1);
                    float v = Bp[(long long)ck * ldb + cn];
                    BsT[n][kk] = (_Float16)((gk < K && gn < N) ? v : 0.0f);
                }
            }
        }

        // speculative prefetch of next K slice (global_prefetch_b8)
        if (k0 + 64 < K) {
            __builtin_prefetch(Ap + (long long)(mBase + (tid >> 1)) * lda + (k0 + 64), 0, 0);
            __builtin_prefetch(Bp + (long long)(tid >> 1) * ldb + (transB ? (k0 + 64) : nBase), 0, 0);
        }
        __syncthreads();

        // ---------------- 2 WMMA K-slices of 32 ----------------
#pragma unroll
        for (int ks = 0; ks < 64; ks += 32) {
            v16h afr[2], bfr[2];
#pragma unroll
            for (int i = 0; i < 2; ++i) {
                int mrow = mW + 16 * i + l16;
                h8 lo = *(const h8*)&As[mrow][ks + aKoff];
                h8 hi = *(const h8*)&As[mrow][ks + 16 + aKoff];
#pragma unroll
                for (int e = 0; e < 8; ++e) {
                    afr[i][e]     = lo[e];
                    afr[i][8 + e] = hi[e];
                }
            }
#pragma unroll
            for (int j = 0; j < 2; ++j) {
                int ncol = nW + 16 * j + l16;
                h8 lo = *(const h8*)&BsT[ncol][ks + bKoff];
                h8 hi = *(const h8*)&BsT[ncol][ks + 16 + bKoff];
#pragma unroll
                for (int e = 0; e < 8; ++e) {
                    bfr[j][e]     = lo[e];
                    bfr[j][8 + e] = hi[e];
                }
            }
#pragma unroll
            for (int i = 0; i < 2; ++i)
#pragma unroll
                for (int j = 0; j < 2; ++j)
                    acc[i][j] = __builtin_amdgcn_wmma_f32_16x16x32_f16(
                        false, afr[i], false, bfr[j], (short)0, acc[i][j], false, false);
        }
        __syncthreads();
    }

    // epilogue: alpha scale, bias, activation, bounds-checked store
#pragma unroll
    for (int i = 0; i < 2; ++i)
#pragma unroll
        for (int j = 0; j < 2; ++j)
#pragma unroll
            for (int r = 0; r < 8; ++r) {
                int row = mBase + mW + 16 * i + ((lane < 16) ? r : r + 8);
                int col = nBase + nW + 16 * j + l16;
                if (row < M && col < N) {
                    float v = acc[i][j][r] * alpha;
                    if (bias) v += bias[col];
                    if (act == 1) {               // gelu (tanh approx)
                        float x = v;
                        float t = tanhf(0.7978845608f * (x + 0.044715f * x * x * x));
                        v = 0.5f * x * (1.0f + t);
                    } else if (act == 2) {        // tanh
                        v = tanhf(v);
                    }
                    Cp[(long long)row * ldc + col] = v;
                }
            }
}

// ---------------------------------------------------------------------------
// LayerNorm with optional residual add, batch-strided in/out (for concat dsts)
// ---------------------------------------------------------------------------
__global__ __launch_bounds__(256)
void layernorm_kernel(float* __restrict__ out, const float* __restrict__ x,
                      const float* __restrict__ resid,
                      const float* __restrict__ g, const float* __restrict__ beta,
                      int S, long long outBS, long long xBS, long long rBS)
{
    const int D = 768;
    int row = blockIdx.x;
    int bb = row / S, ss = row - bb * S;
    const float* xp = x + (long long)bb * xBS + (long long)ss * D;
    const float* rp = resid ? resid + (long long)bb * rBS + (long long)ss * D : nullptr;
    float*       op = out + (long long)bb * outBS + (long long)ss * D;
    __shared__ float red[256];

    float s1 = 0.0f;
    for (int d = threadIdx.x; d < D; d += 256)
        s1 += xp[d] + (rp ? rp[d] : 0.0f);
    red[threadIdx.x] = s1; __syncthreads();
    for (int st = 128; st > 0; st >>= 1) {
        if (threadIdx.x < st) red[threadIdx.x] += red[threadIdx.x + st];
        __syncthreads();
    }
    float mean = red[0] / (float)D;
    __syncthreads();

    float s2 = 0.0f;
    for (int d = threadIdx.x; d < D; d += 256) {
        float v = xp[d] + (rp ? rp[d] : 0.0f) - mean;
        s2 += v * v;
    }
    red[threadIdx.x] = s2; __syncthreads();
    for (int st = 128; st > 0; st >>= 1) {
        if (threadIdx.x < st) red[threadIdx.x] += red[threadIdx.x + st];
        __syncthreads();
    }
    float rstd = rsqrtf(red[0] / (float)D + 1e-5f);

    for (int d = threadIdx.x; d < D; d += 256) {
        float v = xp[d] + (rp ? rp[d] : 0.0f);
        op[d] = (v - mean) * rstd * g[d] + beta[d];
    }
}

// ---------------------------------------------------------------------------
// In-place masked softmax over last dim. mask is per-(b, key) or null.
// ---------------------------------------------------------------------------
__global__ __launch_bounds__(256)
void softmax_kernel(float* __restrict__ sc, const float* __restrict__ mask,
                    int rowsPerB, int Sk)
{
    long long row = blockIdx.x;
    int bb = (int)(row / rowsPerB);
    float* p = sc + row * (long long)Sk;
    const float* mp = mask ? mask + (long long)bb * Sk : nullptr;
    __shared__ float red[256];

    float mx = -1e30f;
    for (int k = threadIdx.x; k < Sk; k += 256) {
        float v = p[k] + (mp ? (1.0f - mp[k]) * -10000.0f : 0.0f);
        mx = fmaxf(mx, v);
    }
    red[threadIdx.x] = mx; __syncthreads();
    for (int st = 128; st > 0; st >>= 1) {
        if (threadIdx.x < st) red[threadIdx.x] = fmaxf(red[threadIdx.x], red[threadIdx.x + st]);
        __syncthreads();
    }
    mx = red[0]; __syncthreads();

    float sum = 0.0f;
    for (int k = threadIdx.x; k < Sk; k += 256) {
        float v = p[k] + (mp ? (1.0f - mp[k]) * -10000.0f : 0.0f);
        float e = __expf(v - mx);
        p[k] = e; sum += e;
    }
    red[threadIdx.x] = sum; __syncthreads();
    for (int st = 128; st > 0; st >>= 1) {
        if (threadIdx.x < st) red[threadIdx.x] += red[threadIdx.x + st];
        __syncthreads();
    }
    float inv = 1.0f / red[0];
    for (int k = threadIdx.x; k < Sk; k += 256) p[k] *= inv;
}

// ---------------------------------------------------------------------------
// Ragged per-utterance text extraction (roberta slicing rule). One block per b.
// ---------------------------------------------------------------------------
__global__ __launch_bounds__(256)
void gather_text_kernel(const float* __restrict__ text_lin,
                        const int* __restrict__ sep, const int* __restrict__ utt,
                        float* __restrict__ feat, float* __restrict__ tmask)
{
    int b = blockIdx.x;
    __shared__ int s_start, s_len;
    if (threadIdx.x == 0) {
        int u = utt[b];
        int cnt = 0, pre = 0, cur = 0;
        for (int t = 0; t < 512; ++t) {
            if (sep[b * 512 + t] != 0) {
                ++cnt;
                if (cnt == u) pre = t;
                if (cnt == u + 1) { cur = t; break; }
            }
        }
        int start = (u == 0) ? 1 : pre + 2;
        int len   = (u == 0) ? (cur - 1) : (cur - pre - 2);
        if (len > 128) len = 128;
        if (len < 0) len = 0;
        s_start = start; s_len = len;
    }
    __syncthreads();
    int start = s_start, len = s_len;
    for (int idx = threadIdx.x; idx < 128 * 768; idx += blockDim.x) {
        int j = idx / 768, d = idx - j * 768;
        int gidx = start + j;
        if (gidx < 0) gidx = 0;
        if (gidx > 511) gidx = 511;
        float v = (j < len) ? text_lin[((long long)b * 512 + gidx) * 768 + d] : 0.0f;
        feat[((long long)b * 128 + j) * 768 + d] = v;
    }
    for (int j = threadIdx.x; j < 128; j += blockDim.x)
        tmask[b * 128 + j] = (j < len) ? 1.0f : 0.0f;
}

__global__ void add_kernel(float* __restrict__ out, const float* __restrict__ a,
                           const float* __restrict__ b, long long n)
{
    long long i = (long long)blockIdx.x * blockDim.x + threadIdx.x;
    long long st = (long long)gridDim.x * blockDim.x;
    for (; i < n; i += st) out[i] = a[i] + b[i];
}

__global__ void addpos_kernel(float* __restrict__ x, const float* __restrict__ pos,
                              int S)
{
    long long n = 16LL * S * 768;
    long long i = (long long)blockIdx.x * blockDim.x + threadIdx.x;
    long long st = (long long)gridDim.x * blockDim.x;
    for (; i < n; i += st) {
        long long sd = i % ((long long)S * 768);
        x[i] += pos[sd];
    }
}

__global__ void fmask_kernel(const float* __restrict__ tmask, const float* __restrict__ amask,
                             const float* __restrict__ vmask, float* __restrict__ fmask)
{
    for (int i = blockIdx.x * blockDim.x + threadIdx.x; i < 16 * 896;
         i += gridDim.x * blockDim.x) {
        int b = i / 896, s = i - b * 896;
        float v;
        if (s < 128)      v = tmask[b * 128 + s];
        else if (s < 640) v = amask[b * 512 + (s - 128)];
        else              v = vmask[b * 256 + (s - 640)];
        fmask[i] = v;
    }
}

__global__ __launch_bounds__(256)
void dotv_kernel(const float* __restrict__ tb, const float* __restrict__ v,
                 float* __restrict__ s)
{
    int row = blockIdx.x;  // 16*896 rows
    __shared__ float red[256];
    float acc = 0.0f;
    for (int d = threadIdx.x; d < 768; d += 256)
        acc += tb[(long long)row * 768 + d] * v[d];
    red[threadIdx.x] = acc; __syncthreads();
    for (int st = 128; st > 0; st >>= 1) {
        if (threadIdx.x < st) red[threadIdx.x] += red[threadIdx.x + st];
        __syncthreads();
    }
    if (threadIdx.x == 0) s[row] = red[0];
}

__global__ void pool_kernel(const float* __restrict__ att, const float* __restrict__ fin,
                            float* __restrict__ pooled)
{
    int b = blockIdx.x;
    int d = blockIdx.y * blockDim.x + threadIdx.x;
    if (d >= 768) return;
    float acc = 0.0f;
    for (int s = 0; s < 896; ++s)
        acc += att[b * 896 + s] * fin[((long long)b * 896 + s) * 768 + d];
    pooled[b * 768 + d] = acc;
}

// ---------------------------------------------------------------------------
// Host-side parameter structs (JAX pytree sorted-key flatten order)
// ---------------------------------------------------------------------------
struct AttnP { const float *bk,*bo,*bq,*bv,*wk,*wo,*wq,*wv; };
struct FfnP  { const float *b1,*b2,*w1,*w2; };
struct EncLayerP { AttnP attn; FfnP ffn; const float *ln1b,*ln1g,*ln2b,*ln2g; };
struct EncP  { EncLayerP layers[3]; const float* pos; };
struct XLayerP { AttnP attn; FfnP ffn; const float *ln2b,*ln2g,*lnkb,*lnkg,*lnqb,*lnqg; };
struct CrossP { XLayerP layers[4]; const float *lnfb,*lnfg; };

extern "C" void kernel_launch(void* const* d_in, const int* in_sizes, int n_in,
                              void* d_out, int out_size, void* d_ws, size_t ws_size,
                              hipStream_t stream)
{
    (void)in_sizes; (void)n_in; (void)out_size;
    const int Bb = 16, Dd = 768, Hh = 8, dh = 96, FFNd = 3072;

    // ---- inputs (setup_inputs insertion order) ----
    const float* text_hidden = (const float*)d_in[0];   // [16,512,1024]
    const int*   sep_mask    = (const int*)  d_in[1];   // [16,512]
    const int*   utt_idx     = (const int*)  d_in[2];   // [16]
    const float* audio_in    = (const float*)d_in[3];   // [16,512,512]
    const float* audio_mask  = (const float*)d_in[4];   // [16,512]
    const float* vision_in   = (const float*)d_in[5];   // [16,256,775]
    const float* vision_mask = (const float*)d_in[6];   // [16,256]

    // ---- params: pytree sorted-key flatten order, starting at d_in[7] ----
    int pi = 7;
    auto nx = [&]() { return (const float*)d_in[pi++]; };
    auto readAttn = [&](AttnP& a) {
        a.bk=nx(); a.bo=nx(); a.bq=nx(); a.bv=nx();
        a.wk=nx(); a.wo=nx(); a.wq=nx(); a.wv=nx();
    };
    auto readFfn = [&](FfnP& f) { f.b1=nx(); f.b2=nx(); f.w1=nx(); f.w2=nx(); };
    auto readEnc = [&](EncP& e) {
        for (int l = 0; l < 3; ++l) {
            readAttn(e.layers[l].attn); readFfn(e.layers[l].ffn);
            e.layers[l].ln1b=nx(); e.layers[l].ln1g=nx();
            e.layers[l].ln2b=nx(); e.layers[l].ln2g=nx();
        }
        e.pos = nx();
    };
    auto readCross = [&](CrossP& c) {
        for (int l = 0; l < 4; ++l) {
            readAttn(c.layers[l].attn); readFfn(c.layers[l].ffn);
            c.layers[l].ln2b=nx(); c.layers[l].ln2g=nx();
            c.layers[l].lnkb=nx(); c.layers[l].lnkg=nx();
            c.layers[l].lnqb=nx(); c.layers[l].lnqg=nx();
        }
        c.lnfb=nx(); c.lnfg=nx();
    };

    const float *add_b = nx(), *add_v = nx(), *add_w = nx();      // addatt: b,v,w
    EncP aenc;  readEnc(aenc);                                    // audio_enc
    const float *al_b = nx(), *al_w = nx();                       // audio_linear
    const float *cl_b = nx(), *cl_w = nx();                       // classifier
    CrossP ta;  readCross(ta);                                    // ta
    CrossP tav; readCross(tav);                                   // tav
    const float *tl_b = nx(), *tl_w = nx();                       // text_linear
    EncP venc;  readEnc(venc);                                    // vision_enc
    const float *vl_b = nx(), *vl_w = nx();                       // vision_linear

    // ---- workspace bump allocator ----
    float* base = (float*)d_ws;
    size_t off = 0;
    auto alloc = [&](size_t nflt) {
        float* p = base + off;
        off += (nflt + 255) & ~(size_t)255;
        return p;
    };
    float* TEXT_LIN  = alloc(16LL*512*768);
    float* TEXT_FEAT = alloc(16LL*128*768);
    float* TEXT_MASK = alloc(16LL*128);
    float* A_X   = alloc(16LL*512*768);
    float* V_X   = alloc(16LL*256*768);
    float* TA    = alloc(16LL*640*768);
    float* FINAL = alloc(16LL*896*768);
    float* XWORK = alloc(16LL*640*768);
    float* XQ    = alloc(16LL*640*768);
    float* YK    = alloc(16LL*640*768);
    float* TMP   = alloc(16LL*640*768);
    float* WQ    = alloc(16LL*896*768);   // also reused as tanh buffer
    float* WK    = alloc(16LL*640*768);
    float* WV    = alloc(16LL*640*768);
    float* WC    = alloc(16LL*640*768);
    float* SCR   = alloc(16LL*8*512*512); // scores / ffn hidden (shared)
    float* SVEC  = alloc(16LL*896);
    float* FMASK = alloc(16LL*896);
    float* POOLED= alloc(16LL*768);
    if (off * sizeof(float) > ws_size) return;   // workspace too small: bail

    // ---- launch helpers ----
    auto gemm = [&](const float* A, const float* Bm, const float* bias, float* C,
                    int M, int N, int K, int batches, int nheads,
                    long long sAb, long long sAh, int lda,
                    long long sBb, long long sBh, int ldb, int transB,
                    long long sCb, long long sCh, int ldc,
                    float alpha, int act) {
        dim3 grid((N + 63) / 64, (M + 63) / 64, batches * nheads);
        gemm_wmma_kernel<<<grid, dim3(128), 0, stream>>>(
            A, Bm, bias, C, M, N, K, nheads,
            sAb, sAh, lda, sBb, sBh, ldb, transB, sCb, sCh, ldc, alpha, act);
    };
    auto gemmS = [&](const float* A, const float* W, const float* bias, float* C,
                     int M, int N, int K, int act) {
        gemm(A, W, bias, C, M, N, K, 1, 1, 0, 0, K, 0, 0, N, 0, 0, 0, N, 1.0f, act);
    };
    auto ln = [&](float* out, long long outBS, const float* x, long long xBS,
                  const float* resid, long long rBS,
                  const float* g, const float* b, int S) {
        layernorm_kernel<<<dim3(16 * S), dim3(256), 0, stream>>>(
            out, x, resid, g, b, S, outBS, xBS, rBS);
    };
    auto softmax = [&](float* sc, const float* mask, int rowsPerB, int Sk, long long rows) {
        softmax_kernel<<<dim3((unsigned)rows), dim3(256), 0, stream>>>(sc, mask, rowsPerB, Sk);
    };
    auto addv = [&](float* out, const float* a, const float* b, long long n) {
        int blocks = (int)((n + 255) / 256); if (blocks > 4096) blocks = 4096;
        add_kernel<<<dim3(blocks), dim3(256), 0, stream>>>(out, a, b, n);
    };

    auto mha = [&](float* out, const float* qsrc, int Sq, const float* kvsrc, int Skv,
                   const AttnP& p, const float* keymask) {
        gemmS(qsrc,  p.wq, p.bq, WQ, Bb * Sq,  Dd, Dd, 0);
        gemmS(kvsrc, p.wk, p.bk, WK, Bb * Skv, Dd, Dd, 0);
        gemmS(kvsrc, p.wv, p.bv, WV, Bb * Skv, Dd, Dd, 0);
        // scores[b,h,q,k] = (Q . K^T) / sqrt(dh)
        gemm(WQ, WK, nullptr, SCR, Sq, Skv, dh, Bb, Hh,
             (long long)Sq * Dd, dh, Dd,
             (long long)Skv * Dd, dh, Dd, 1,
             (long long)Hh * Sq * Skv, (long long)Sq * Skv, Skv,
             0.1020620726f, 0);
        softmax(SCR, keymask, Hh * Sq, Skv, (long long)Bb * Hh * Sq);
        // ctx[b,q,h*dh] = probs @ V
        gemm(SCR, WV, nullptr, WC, Sq, dh, Skv, Bb, Hh,
             (long long)Hh * Sq * Skv, (long long)Sq * Skv, Skv,
             (long long)Skv * Dd, dh, Dd, 0,
             (long long)Sq * Dd, dh, Dd,
             1.0f, 0);
        gemmS(WC, p.wo, p.bo, out, Bb * Sq, Dd, Dd, 0);
    };

    auto self_enc = [&](float* xbuf, int S, const EncP& e, const float* kmask) {
        addpos_kernel<<<dim3(2048), dim3(256), 0, stream>>>(xbuf, e.pos, S);
        long long rowBS = (long long)S * Dd;
        for (int l = 0; l < 3; ++l) {
            const EncLayerP& L = e.layers[l];
            mha(TMP, xbuf, S, xbuf, S, L.attn, kmask);
            ln(xbuf, rowBS, TMP, rowBS, xbuf, rowBS, L.ln1g, L.ln1b, S);
            gemmS(xbuf, L.ffn.w1, L.ffn.b1, SCR, Bb * S, FFNd, Dd, 1);
            gemmS(SCR,  L.ffn.w2, L.ffn.b2, TMP, Bb * S, Dd, FFNd, 0);
            ln(xbuf, rowBS, TMP, rowBS, xbuf, rowBS, L.ln2g, L.ln2b, S);
        }
    };

    auto cross_enc = [&](float* xbuf, int Sx, const float* ybuf, int Sy,
                         const CrossP& c, float* dest, long long destBS) {
        long long xBS = (long long)Sx * Dd, yBS = (long long)Sy * Dd;
        for (int l = 0; l < 4; ++l) {
            const XLayerP& L = c.layers[l];
            ln(XQ, xBS, xbuf, xBS, nullptr, 0, L.lnqg, L.lnqb, Sx);
            ln(YK, yBS, ybuf, yBS, nullptr, 0, L.lnkg, L.lnkb, Sy);
            mha(TMP, XQ, Sx, YK, Sy, L.attn, nullptr);
            addv(xbuf, xbuf, TMP, (long long)Bb * Sx * Dd);
            ln(XQ, xBS, xbuf, xBS, nullptr, 0, L.ln2g, L.ln2b, Sx);
            gemmS(XQ,  L.ffn.w1, L.ffn.b1, SCR, Bb * Sx, FFNd, Dd, 1);
            gemmS(SCR, L.ffn.w2, L.ffn.b2, TMP, Bb * Sx, Dd, FFNd, 0);
            addv(xbuf, xbuf, TMP, (long long)Bb * Sx * Dd);
        }
        ln(dest, destBS, xbuf, xBS, nullptr, 0, c.lnfg, c.lnfb, Sx);
    };

    // =======================================================================
    // Forward pass
    // =======================================================================
    gemmS(text_hidden, tl_w, tl_b, TEXT_LIN, Bb * 512, Dd, 1024, 0);
    gather_text_kernel<<<dim3(16), dim3(256), 0, stream>>>(
        TEXT_LIN, sep_mask, utt_idx, TEXT_FEAT, TEXT_MASK);

    // audio branch
    gemmS(audio_in, al_w, al_b, A_X, Bb * 512, Dd, 512, 0);
    self_enc(A_X, 512, aenc, audio_mask);

    // vision branch (DV = 775)
    gemmS(vision_in, vl_w, vl_b, V_X, Bb * 256, Dd, 775, 0);
    self_enc(V_X, 256, venc, vision_mask);

    // cross T<->A (shared module), writing directly into concat layout TA[16,640,768]
    hipMemcpyAsync(XWORK, TEXT_FEAT, 16LL*128*768*sizeof(float),
                   hipMemcpyDeviceToDevice, stream);
    cross_enc(XWORK, 128, A_X, 512, ta, TA, 640LL * Dd);                    // rows 0..127
    hipMemcpyAsync(XWORK, A_X, 16LL*512*768*sizeof(float),
                   hipMemcpyDeviceToDevice, stream);
    cross_enc(XWORK, 512, TEXT_FEAT, 128, ta, TA + 128LL * Dd, 640LL * Dd); // rows 128..639

    // cross TA<->V into FINAL[16,896,768]
    hipMemcpyAsync(XWORK, TA, 16LL*640*768*sizeof(float),
                   hipMemcpyDeviceToDevice, stream);
    cross_enc(XWORK, 640, V_X, 256, tav, FINAL, 896LL * Dd);                // rows 0..639
    hipMemcpyAsync(XWORK, V_X, 16LL*256*768*sizeof(float),
                   hipMemcpyDeviceToDevice, stream);
    cross_enc(XWORK, 256, TA, 640, tav, FINAL + 640LL * Dd, 896LL * Dd);    // rows 640..895

    // additive attention pooling
    fmask_kernel<<<dim3(56), dim3(256), 0, stream>>>(TEXT_MASK, audio_mask, vision_mask, FMASK);
    gemmS(FINAL, add_w, add_b, WQ, Bb * 896, Dd, Dd, 2);                    // tanh(final@w+b)
    dotv_kernel<<<dim3(16 * 896), dim3(256), 0, stream>>>(WQ, add_v, SVEC);
    softmax(SVEC, FMASK, 1, 896, 16);
    pool_kernel<<<dim3(16, 6), dim3(128), 0, stream>>>(SVEC, FINAL, POOLED);

    // classifier [16,768] @ [768,7] -> d_out
    gemmS(POOLED, cl_w, cl_b, (float*)d_out, Bb, 7, Dd, 0);
}
